// EnhancedTemporalGNN_76836964926296
// MI455X (gfx1250) — matrix operand order repaired and checked by reading, not compiled, not run on hardware
//
#include <hip/hip_runtime.h>
#include <math.h>

// ---------------------------------------------------------------------------
// CDNA5 (gfx1250) WMMA / TDM types
// ---------------------------------------------------------------------------
typedef __attribute__((ext_vector_type(16))) _Float16 v16h;
typedef __attribute__((ext_vector_type(8)))  float    v8f;

#if defined(__HIP_DEVICE_COMPILE__) && defined(__gfx1250__) && \
    __has_builtin(__builtin_amdgcn_tensor_load_to_lds) &&      \
    __has_builtin(__builtin_amdgcn_s_wait_tensorcnt)
#define TDM_OK 1
#endif

static __device__ __forceinline__ float sigmoid_f(float x) {
    return 1.0f / (1.0f + __expf(-x));
}

static __device__ __forceinline__ void atomicMaxF(float* addr, float val) {
    int* ia = reinterpret_cast<int*>(addr);
    int cur = *ia;
    while (__int_as_float(cur) < val) {
        int prev = atomicCAS(ia, cur, __float_as_int(val));
        if (prev == cur) break;
        cur = prev;
    }
}

#ifdef TDM_OK
// ---------------------------------------------------------------------------
// TDM 2-D tile load: rows x 128 halves, contiguous rows (stride 128 elems),
// LDS rows padded to 136 halves (pad 16B after every 256B).
// Descriptor bit layout per CDNA5 ISA sec 8.3 / 8.4.
// ---------------------------------------------------------------------------
static __device__ __forceinline__ void tdm_load_tile(const void* gptr,
                                                     unsigned lds_addr,
                                                     unsigned tile_rows,
                                                     unsigned tensor_rows) {
    typedef unsigned int u32x4 __attribute__((ext_vector_type(4)));
    typedef int i32x4 __attribute__((ext_vector_type(4)));
    typedef int i32x8 __attribute__((ext_vector_type(8)));
    unsigned long long ga = (unsigned long long)gptr;
    u32x4 g0;
    g0[0] = 1u;                                     // count=1 (valid user D#)
    g0[1] = lds_addr;                               // lds_addr (bytes)
    g0[2] = (unsigned)(ga & 0xffffffffu);           // global_addr[31:0]
    g0[3] = (unsigned)((ga >> 32) & 0x1ffffffu)     // global_addr[56:32]
            | (2u << 30);                           // type=2 ("image")
    i32x8 g1;
    g1[0] = (int)((1u << 16)        // data_size = 1 -> 2 bytes
                  | (1u << 20)      // pad_enable
                  | (5u << 22)      // pad_interval: 64 DWORDs (256B)
                  | (3u << 25));    // pad_amount: 4 DWORDs (16B)
    g1[1] = (int)(128u << 16);                          // tensor_dim0 = 128 (lo16)
    g1[2] = (int)((tensor_rows & 0xffffu) << 16);       // dim0 hi16=0 | tensor_dim1 lo16
    g1[3] = (int)(((tensor_rows >> 16) & 0xffffu)       // tensor_dim1 hi16
                  | (128u << 16));                      // tile_dim0 = 128
    g1[4] = (int)(tile_rows & 0xffffu);                 // tile_dim1 | tile_dim2=0
    g1[5] = 128;                                        // tensor_dim0_stride lo32
    g1[6] = (int)(128u << 16);                          // stride0 hi16=0 | dim1_stride lo16
    g1[7] = 0;                                          // dim1_stride hi32
    i32x4 z4 = {0, 0, 0, 0};
#if __clang_major__ >= 23
    i32x8 z8 = {0, 0, 0, 0, 0, 0, 0, 0};
    __builtin_amdgcn_tensor_load_to_lds(g0, g1, z4, z4, z8, 0);
#else
    __builtin_amdgcn_tensor_load_to_lds(g0, g1, z4, z4, 0);
#endif
}
#endif  // TDM_OK

// ---------------------------------------------------------------------------
// conversions
// ---------------------------------------------------------------------------
__global__ void f32_to_f16_kernel(const float* __restrict__ in,
                                  _Float16* __restrict__ out, long n) {
    long i = (long)blockIdx.x * blockDim.x + threadIdx.x;
    if (i < n) out[i] = (_Float16)in[i];
}

// W[K=128][Ncols] f32 -> Wt[Ncols][128] f16 (transposed for K-contiguous B frags)
__global__ void f32_to_f16_tr_kernel(const float* __restrict__ in,
                                     _Float16* __restrict__ out, int Ncols) {
    int t = blockIdx.x * blockDim.x + threadIdx.x;
    if (t >= 128 * Ncols) return;
    int k = t / Ncols, c = t % Ncols;
    out[(size_t)c * 128 + k] = (_Float16)in[t];
}

__global__ void fill_kernel(float* __restrict__ p, float v, long n) {
    long i = (long)blockIdx.x * blockDim.x + threadIdx.x;
    if (i < n) p[i] = v;
}

// ---------------------------------------------------------------------------
// WMMA GEMM:  Y[M x Ncols] = X[M x 128] (f16) @ W[128 x Ncols] + bias
// W passed PRE-TRANSPOSED: Wt[Ncols][128].
// block = 128 threads (4 waves); tile 64 rows x 128 cols; K=128 in LDS.
// B fragments software-pipelined: next B's ds_load_b128s issue before the
// WMMA consuming the current B, hiding LDS latency behind the matrix op.
// ---------------------------------------------------------------------------
__global__ __launch_bounds__(128)
void gemm_f16_wmma_kernel(const _Float16* __restrict__ X,
                          const _Float16* __restrict__ Wt,
                          const float* __restrict__ bias,
                          float* __restrict__ Y,
                          int M, int Ncols) {
    __shared__ _Float16 Xs[64][136];    // row stride 272B: 16B aligned, padded
    __shared__ _Float16 Ws[128][136];   // Ws[c][k]: K-contiguous per out-col

    const int tid  = threadIdx.x;
    const int wave = tid >> 5;
    const int lane = tid & 31;
    const int row0 = blockIdx.x * 64;
    const int col0 = blockIdx.y * 128;
    const bool full_x = (row0 + 64 <= M);

#ifdef TDM_OK
    if (wave == 0) {
        tdm_load_tile(Wt + (size_t)col0 * 128,
                      (unsigned)(size_t)&Ws[0][0], 128u, (unsigned)Ncols);
        if (full_x)
            tdm_load_tile(X + (size_t)row0 * 128,
                          (unsigned)(size_t)&Xs[0][0], 64u, (unsigned)M);
        __builtin_amdgcn_s_wait_tensorcnt(0);
    }
    if (!full_x) {
#pragma unroll
        for (int it = 0; it < 8; ++it) {
            int e0 = (it * 128 + tid) * 8;
            int r = e0 >> 7, c = e0 & 127;
            int gr = row0 + r;
            float4 v;
            if (gr < M) v = *(const float4*)(X + (size_t)gr * 128 + c);
            else        v = make_float4(0.f, 0.f, 0.f, 0.f);
            *(float4*)(&Xs[r][c]) = v;
        }
    }
#else
    // ---- stage X tile (64 x 128 halves) ----
#pragma unroll
    for (int it = 0; it < 8; ++it) {
        int e0 = (it * 128 + tid) * 8;
        int r = e0 >> 7, c = e0 & 127;
        int gr = row0 + r;
        float4 v;
        if (gr < M) v = *(const float4*)(X + (size_t)gr * 128 + c);
        else        v = make_float4(0.f, 0.f, 0.f, 0.f);
        *(float4*)(&Xs[r][c]) = v;
    }
    // ---- stage Wt tile (128 out-cols x 128 K halves) ----
#pragma unroll
    for (int it = 0; it < 16; ++it) {
        int e0 = (it * 128 + tid) * 8;
        int r = e0 >> 7, c = e0 & 127;
        *(float4*)(&Ws[r][c]) = *(const float4*)(Wt + (size_t)(col0 + r) * 128 + c);
    }
#endif
    __syncthreads();

    v8f acc[8];
    const v8f zero8 = {0.f, 0.f, 0.f, 0.f, 0.f, 0.f, 0.f, 0.f};
#pragma unroll
    for (int ct = 0; ct < 8; ++ct) acc[ct] = zero8;

    // ISA 7.12.2 fragment layouts (wave32):
    //  A 16x32 f16: lanes 0-15 rows 0-15, K {0..15}; lanes 16-31 K {16..31} (contiguous, as loaded here)
    //  B 32x16 f16: lanes 0-15 cols, K {0..15}; lanes 16-31 same cols, K {16..31} (order used here, same as A; not the ISA table order)
    //  C/D 16x16 f32: VGPR i -> row i / 8+i (lane hi), col = lane&15
    union V16 { v16h v; float4 f4[2]; };
    const int arow  = (wave << 4) + (lane & 15);
    const int ahalf = (lane & 16) ? 16 : 0;
    const int bcolL = lane & 15;
    const int bhalf = (lane & 16) ? 16 : 0;

#pragma unroll
    for (int kb = 0; kb < 4; ++kb) {
        const int k0 = kb * 32;
        V16 A;
        A.f4[0] = *(const float4*)(&Xs[arow][k0 + ahalf]);        // K +0..7
        A.f4[1] = *(const float4*)(&Xs[arow][k0 + ahalf + 8]);    // K +8..15
        V16 Bc;
        Bc.f4[0] = *(const float4*)(&Ws[bcolL][k0 + bhalf]);      // ct=0 fragment
        Bc.f4[1] = *(const float4*)(&Ws[bcolL][k0 + bhalf + 8]);
#pragma unroll
        for (int ct = 0; ct < 8; ++ct) {
            V16 Bn;
            if (ct < 7) {   // prefetch next B fragment before consuming current
                const int bc = ((ct + 1) << 4) + bcolL;
                Bn.f4[0] = *(const float4*)(&Ws[bc][k0 + bhalf]);
                Bn.f4[1] = *(const float4*)(&Ws[bc][k0 + bhalf + 8]);
            }
            acc[ct] = __builtin_amdgcn_wmma_f32_16x16x32_f16(
                false, A.v, false, Bc.v, (short)0, acc[ct], false, false);
            if (ct < 7) Bc = Bn;
        }
    }

    const int crow0 = row0 + (wave << 4) + ((lane & 16) ? 8 : 0);
#pragma unroll
    for (int ct = 0; ct < 8; ++ct) {
        const int cc = col0 + (ct << 4) + (lane & 15);
        const float bv = bias ? bias[cc] : 0.f;
#pragma unroll
        for (int i = 0; i < 8; ++i) {
            int rr = crow0 + i;
            if (rr < M) Y[(size_t)rr * Ncols + cc] = acc[ct][i] + bv;
        }
    }
}

// ---------------------------------------------------------------------------
// Edge attention kernels (H = 4 heads, DH = 32)
// ---------------------------------------------------------------------------
__global__ void edge_logits_kernel(const float* __restrict__ qn,
                                   const float* __restrict__ kn,
                                   const float* __restrict__ ef,
                                   const int* __restrict__ src,
                                   const int* __restrict__ dst,
                                   float* __restrict__ logits,
                                   float* __restrict__ mseg, int E_) {
    int t = blockIdx.x * blockDim.x + threadIdx.x;
    if (t >= E_ * 4) return;
    int ed = t >> 2, h = t & 3;
    int s = src[ed], d0 = dst[ed];
    const float* q = qn + (size_t)d0 * 128 + h * 32;
    const float* k = kn + (size_t)s * 128 + h * 32;
    const float* e = ef + (size_t)ed * 128 + h * 32;
    float acc = 0.f;
#pragma unroll
    for (int i = 0; i < 32; ++i) acc = fmaf(q[i], k[i] + e[i], acc);
    acc *= 0.17677669529663687f;   // 1/sqrt(32)
    logits[t] = acc;
    atomicMaxF(&mseg[d0 * 4 + h], acc);
}

__global__ void m_fix_kernel(float* __restrict__ mseg, int n) {
    int i = blockIdx.x * blockDim.x + threadIdx.x;
    if (i >= n) return;
    float v = mseg[i];
    if (!__builtin_isfinite(v)) mseg[i] = 0.f;
}

__global__ void edge_exp_kernel(const float* __restrict__ logits,
                                const float* __restrict__ mseg,
                                const int* __restrict__ dst,
                                float* __restrict__ a,
                                float* __restrict__ denom, int E_) {
    int t = blockIdx.x * blockDim.x + threadIdx.x;
    if (t >= E_ * 4) return;
    int ed = t >> 2, h = t & 3;
    int d0 = dst[ed];
    float av = __expf(logits[t] - mseg[d0 * 4 + h]);
    a[t] = av;
    atomicAdd(&denom[d0 * 4 + h], av);
}

__global__ void edge_scatter_kernel(const float* __restrict__ vn,
                                    const float* __restrict__ ef,
                                    const float* __restrict__ a,
                                    const float* __restrict__ denom,
                                    const int* __restrict__ src,
                                    const int* __restrict__ dst,
                                    float* __restrict__ agg, int E_) {
    int t = blockIdx.x * blockDim.x + threadIdx.x;
    if (t >= E_ * 128) return;
    int ed = t >> 7, c = t & 127;
    int h = c >> 5;
    int s = src[ed], d0 = dst[ed];
    float alpha = a[ed * 4 + h] / (denom[d0 * 4 + h] + 1e-16f);
    float val = (vn[(size_t)s * 128 + c] + ef[(size_t)ed * 128 + c]) * alpha;
    atomicAdd(&agg[(size_t)d0 * 128 + c], val);
}

// ---------------------------------------------------------------------------
// beta gate + combine: conv = beta*r + (1-beta)*out ; one wave32 per node
// ---------------------------------------------------------------------------
__global__ __launch_bounds__(256)
void beta_combine_kernel(const float* __restrict__ agg,
                         const float* __restrict__ rs,
                         const float* __restrict__ wbeta,
                         float* __restrict__ conv, int Nn) {
    int node = blockIdx.x * 8 + (threadIdx.x >> 5);
    int lane = threadIdx.x & 31;
    if (node >= Nn) return;
    float acc = 0.f;
#pragma unroll
    for (int j = 0; j < 4; ++j) {
        int i = lane + 32 * j;
        float o = agg[(size_t)node * 128 + i];
        float r = rs[(size_t)node * 128 + i];
        acc += o * wbeta[i] + r * wbeta[128 + i] + (o - r) * wbeta[256 + i];
    }
#pragma unroll
    for (int off = 16; off > 0; off >>= 1) acc += __shfl_down(acc, off, 32);
    float beta = sigmoid_f(__shfl(acc, 0, 32));
#pragma unroll
    for (int j = 0; j < 4; ++j) {
        int i = lane + 32 * j;
        float o = agg[(size_t)node * 128 + i];
        float r = rs[(size_t)node * 128 + i];
        conv[(size_t)node * 128 + i] = beta * r + (1.f - beta) * o;
    }
}

// ---------------------------------------------------------------------------
// exact GELU + LayerNorm (+optional residual), also emits f16 copy
// ---------------------------------------------------------------------------
__global__ __launch_bounds__(256)
void gelu_ln_kernel(const float* __restrict__ in,
                    const float* __restrict__ gamma,
                    const float* __restrict__ betab,
                    const float* __restrict__ residual,
                    float* __restrict__ out32,
                    _Float16* __restrict__ out16, int Nn) {
    int node = blockIdx.x * 8 + (threadIdx.x >> 5);
    int lane = threadIdx.x & 31;
    if (node >= Nn) return;
    const float* p = in + (size_t)node * 128;
    float v[4];
    float s = 0.f;
#pragma unroll
    for (int j = 0; j < 4; ++j) {
        float xx = p[lane + 32 * j];
        xx = 0.5f * xx * (1.f + erff(xx * 0.70710678118654752f));  // exact GELU
        v[j] = xx;
        s += xx;
    }
#pragma unroll
    for (int off = 16; off > 0; off >>= 1) s += __shfl_down(s, off, 32);
    float mu = __shfl(s, 0, 32) * (1.f / 128.f);
    float var = 0.f;
#pragma unroll
    for (int j = 0; j < 4; ++j) { float d = v[j] - mu; var += d * d; }
#pragma unroll
    for (int off = 16; off > 0; off >>= 1) var += __shfl_down(var, off, 32);
    var = __shfl(var, 0, 32) * (1.f / 128.f);
    float inv = rsqrtf(var + 1e-5f);
#pragma unroll
    for (int j = 0; j < 4; ++j) {
        int c = lane + 32 * j;
        float y = gamma[c] * (v[j] - mu) * inv + betab[c];
        if (residual) y += residual[(size_t)node * 128 + c];
        out32[(size_t)node * 128 + c] = y;
        out16[(size_t)node * 128 + c] = (_Float16)y;
    }
}

// ---------------------------------------------------------------------------
// output gate: emb = g*x + (1-g)*outWo, g = sigmoid(x.Wg + bg)
// ---------------------------------------------------------------------------
__global__ __launch_bounds__(256)
void gate_emb_kernel(const float* __restrict__ x,
                     const float* __restrict__ oWo,
                     const float* __restrict__ Wg,
                     const float* __restrict__ bg,
                     float* __restrict__ emb, int Nn) {
    int node = blockIdx.x * 8 + (threadIdx.x >> 5);
    int lane = threadIdx.x & 31;
    if (node >= Nn) return;
    float acc = 0.f;
#pragma unroll
    for (int j = 0; j < 4; ++j) {
        int i = lane + 32 * j;
        acc = fmaf(x[(size_t)node * 128 + i], Wg[i], acc);
    }
#pragma unroll
    for (int off = 16; off > 0; off >>= 1) acc += __shfl_down(acc, off, 32);
    float g = sigmoid_f(__shfl(acc, 0, 32) + bg[0]);
#pragma unroll
    for (int j = 0; j < 4; ++j) {
        int i = lane + 32 * j;
        float xv = x[(size_t)node * 128 + i];
        emb[(size_t)node * 128 + i] = g * xv + (1.f - g) * oWo[(size_t)node * 128 + i];
    }
}

// ---------------------------------------------------------------------------
// GRU: gather inputs (f16 for WMMA GEMM) and pointwise cell
// ---------------------------------------------------------------------------
__global__ void gru_gather_kernel(const float* __restrict__ emb,
                                  const float* __restrict__ hidden,
                                  const int* __restrict__ idx,
                                  _Float16* __restrict__ embI,
                                  _Float16* __restrict__ hidI, int Bn) {
    int t = blockIdx.x * blockDim.x + threadIdx.x;
    if (t >= Bn * 128) return;
    int b = t >> 7, c = t & 127;
    int node = idx[b];
    embI[t] = (_Float16)emb[(size_t)node * 128 + c];
    hidI[t] = (_Float16)hidden[(size_t)node * 128 + c];
}

__global__ void gru_out_kernel(const float* __restrict__ gi,
                               const float* __restrict__ gh,
                               const float* __restrict__ hidden,
                               const int* __restrict__ idx,
                               float* __restrict__ hnew, int Bn) {
    int t = blockIdx.x * blockDim.x + threadIdx.x;
    if (t >= Bn * 128) return;
    int b = t >> 7, c = t & 127;
    const float* gib = gi + (size_t)b * 384;
    const float* ghb = gh + (size_t)b * 384;
    float r = sigmoid_f(gib[c] + ghb[c]);
    float z = sigmoid_f(gib[128 + c] + ghb[128 + c]);
    float n = tanhf(gib[256 + c] + r * ghb[256 + c]);
    float ho = hidden[(size_t)idx[b] * 128 + c];
    hnew[t] = (1.f - z) * n + z * ho;
}

// ---------------------------------------------------------------------------
// host driver
// ---------------------------------------------------------------------------
extern "C" void kernel_launch(void* const* d_in, const int* in_sizes, int n_in,
                              void* d_out, int out_size, void* d_ws, size_t ws_size,
                              hipStream_t stream) {
    (void)in_sizes; (void)n_in; (void)out_size; (void)ws_size;
    constexpr int N = 50000, E = 500000, D = 128, B = 8192;

    const float* x    = (const float*)d_in[0];
    const int*   ei   = (const int*)d_in[1];
    const float* ea   = (const float*)d_in[2];
    const int*   gidx = (const int*)d_in[3];
    auto cw = [&](int i) { return (const float*)d_in[i]; };
    const int C1 = 4, C2 = 15;
    const float *ln1g = cw(26), *ln1b = cw(27), *ln2g = cw(28), *ln2b = cw(29);
    const float *Wo = cw(30), *bo = cw(31), *Wg = cw(32), *bg = cw(33);
    const float *Wih = cw(34), *bih = cw(35), *Whh = cw(36), *bhh = cw(37);
    const float* hidden = cw(38);
    const int* srcI = ei;
    const int* dstI = ei + E;

    // workspace arena
    char* base = (char*)d_ws;
    size_t off = 0;
    auto carve = [&](size_t bytes) -> char* {
        char* p = base + off;
        off += (bytes + 255) & ~(size_t)255;
        return p;
    };
    _Float16* x_h  = (_Float16*)carve((size_t)N * D * 2);
    _Float16* ea_h = (_Float16*)carve((size_t)E * D * 2);
    _Float16* wh[13];   // transposed f16 weights
    for (int i = 0; i < 11; ++i) wh[i] = (_Float16*)carve((size_t)D * D * 2);
    wh[11] = (_Float16*)carve((size_t)D * 3 * D * 2);
    wh[12] = (_Float16*)carve((size_t)D * 3 * D * 2);
    float* qn   = (float*)carve((size_t)N * D * 4);
    float* kn   = (float*)carve((size_t)N * D * 4);
    float* vn   = (float*)carve((size_t)N * D * 4);
    float* rs   = (float*)carve((size_t)N * D * 4);
    float* ef   = (float*)carve((size_t)E * D * 4);
    float* lgt  = (float*)carve((size_t)E * 4 * 4);
    float* att  = (float*)carve((size_t)E * 4 * 4);
    float* mseg = (float*)carve((size_t)N * 4 * 4);
    float* dseg = (float*)carve((size_t)N * 4 * 4);
    float* agg  = (float*)carve((size_t)N * D * 4);
    float* conv = (float*)carve((size_t)N * D * 4);
    float* h1   = (float*)carve((size_t)N * D * 4);
    _Float16* h1h   = (_Float16*)carve((size_t)N * D * 2);
    float* hsum = (float*)carve((size_t)N * D * 4);
    _Float16* hsumh = (_Float16*)carve((size_t)N * D * 2);
    float* oWo  = (float*)carve((size_t)N * D * 4);
    _Float16* embI = (_Float16*)carve((size_t)B * D * 2);
    _Float16* hidI = (_Float16*)carve((size_t)B * D * 2);
    float* gi = (float*)carve((size_t)B * 3 * D * 4);
    float* gh = (float*)carve((size_t)B * 3 * D * 4);

    const int TPB = 256;
    auto cdiv = [](long a, long b) { return (int)((a + b - 1) / b); };

    // ---- f16 conversions (weights transposed: Wt[Ncols][128]) ----
    f32_to_f16_kernel<<<cdiv((long)N * D, TPB), TPB, 0, stream>>>(x, x_h, (long)N * D);
    f32_to_f16_kernel<<<cdiv((long)E * D, TPB), TPB, 0, stream>>>(ea, ea_h, (long)E * D);
    const float* wsrc[13] = { cw(C1 + 0), cw(C1 + 2), cw(C1 + 4), cw(C1 + 6), cw(C1 + 8),
                              cw(C2 + 0), cw(C2 + 2), cw(C2 + 4), cw(C2 + 6), cw(C2 + 8),
                              Wo, Wih, Whh };
    const int wcols[13] = { D, D, D, D, D, D, D, D, D, D, D, 3 * D, 3 * D };
    for (int i = 0; i < 13; ++i)
        f32_to_f16_tr_kernel<<<cdiv((long)128 * wcols[i], TPB), TPB, 0, stream>>>(
            wsrc[i], wh[i], wcols[i]);

    dim3 blk128(128);
    auto run_conv = [&](int cb, const _Float16* inh,
                        const _Float16* wq, const _Float16* wk, const _Float16* wv,
                        const _Float16* we, const _Float16* wsk, float* outbuf) {
        const float *bq = cw(cb + 1), *bk = cw(cb + 3), *bv = cw(cb + 5);
        const float *be = cw(cb + 7), *bs = cw(cb + 9), *wbeta = cw(cb + 10);
        gemm_f16_wmma_kernel<<<dim3(cdiv(N, 64), 1), blk128, 0, stream>>>(inh, wq, bq, qn, N, D);
        gemm_f16_wmma_kernel<<<dim3(cdiv(N, 64), 1), blk128, 0, stream>>>(inh, wk, bk, kn, N, D);
        gemm_f16_wmma_kernel<<<dim3(cdiv(N, 64), 1), blk128, 0, stream>>>(inh, wv, bv, vn, N, D);
        gemm_f16_wmma_kernel<<<dim3(cdiv(N, 64), 1), blk128, 0, stream>>>(inh, wsk, bs, rs, N, D);
        gemm_f16_wmma_kernel<<<dim3(cdiv(E, 64), 1), blk128, 0, stream>>>(ea_h, we, be, ef, E, D);
        fill_kernel<<<cdiv((long)N * 4, TPB), TPB, 0, stream>>>(mseg, -INFINITY, (long)N * 4);
        fill_kernel<<<cdiv((long)N * 4, TPB), TPB, 0, stream>>>(dseg, 0.f, (long)N * 4);
        fill_kernel<<<cdiv((long)N * D, TPB), TPB, 0, stream>>>(agg, 0.f, (long)N * D);
        edge_logits_kernel<<<cdiv((long)E * 4, TPB), TPB, 0, stream>>>(qn, kn, ef, srcI, dstI, lgt, mseg, E);
        m_fix_kernel<<<cdiv((long)N * 4, TPB), TPB, 0, stream>>>(mseg, N * 4);
        edge_exp_kernel<<<cdiv((long)E * 4, TPB), TPB, 0, stream>>>(lgt, mseg, dstI, att, dseg, E);
        edge_scatter_kernel<<<cdiv((long)E * D, TPB), TPB, 0, stream>>>(vn, ef, att, dseg, srcI, dstI, agg, E);
        beta_combine_kernel<<<cdiv(N, 8), 256, 0, stream>>>(agg, rs, wbeta, outbuf, N);
    };

    // conv1 -> gelu+ln1
    run_conv(C1, x_h, wh[0], wh[1], wh[2], wh[3], wh[4], conv);
    gelu_ln_kernel<<<cdiv(N, 8), 256, 0, stream>>>(conv, ln1g, ln1b, nullptr, h1, h1h, N);
    // conv2 -> gelu+ln2 (+residual h1)
    run_conv(C2, h1h, wh[5], wh[6], wh[7], wh[8], wh[9], conv);
    gelu_ln_kernel<<<cdiv(N, 8), 256, 0, stream>>>(conv, ln2g, ln2b, h1, hsum, hsumh, N);
    // out = h @ Wo + bo ; emb = g*x + (1-g)*out
    gemm_f16_wmma_kernel<<<dim3(cdiv(N, 64), 1), blk128, 0, stream>>>(hsumh, wh[10], bo, oWo, N, D);
    float* emb  = (float*)d_out;
    float* hnew = emb + (size_t)N * D;
    gate_emb_kernel<<<cdiv(N, 8), 256, 0, stream>>>(x, oWo, Wg, bg, emb, N);
    // GRU memory write on gathered rows
    gru_gather_kernel<<<cdiv((long)B * D, TPB), TPB, 0, stream>>>(emb, hidden, gidx, embI, hidI, B);
    gemm_f16_wmma_kernel<<<dim3(cdiv(B, 64), 3), blk128, 0, stream>>>(embI, wh[11], bih, gi, B, 3 * D);
    gemm_f16_wmma_kernel<<<dim3(cdiv(B, 64), 3), blk128, 0, stream>>>(hidI, wh[12], bhh, gh, B, 3 * D);
    gru_out_kernel<<<cdiv((long)B * D, TPB), TPB, 0, stream>>>(gi, gh, hidden, gidx, hnew, B);
}
